// SGC_77129022701608
// MI455X (gfx1250) — matrix operand order repaired and checked by reading, not compiled, not run on hardware
//
#include <hip/hip_runtime.h>

#define N_NODES 100000
#define N_EDGES 1280000
#define FEAT 64
#define NCLS 40

typedef __attribute__((ext_vector_type(16))) _Float16 v16h;
typedef __attribute__((ext_vector_type(8)))  float    v8f;

__device__ __forceinline__ void atomic_add_f32(float* p, float v) {
    (void)__hip_atomic_fetch_add(p, v, __ATOMIC_RELAXED, __HIP_MEMORY_SCOPE_AGENT);
}

// ---------------- degree / norm ----------------
__global__ void k_init_deg(float* __restrict__ deg) {
    int i = blockIdx.x * blockDim.x + threadIdx.x;
    if (i < N_NODES) deg[i] = 1.0f;   // self loop
}

__global__ void k_accum_deg(const int* __restrict__ dst, float* __restrict__ deg) {
    int e = blockIdx.x * blockDim.x + threadIdx.x;
    if (e < N_EDGES) atomic_add_f32(&deg[dst[e]], 1.0f);
}

__global__ void k_dinv(float* __restrict__ deg) {
    int i = blockIdx.x * blockDim.x + threadIdx.x;
    if (i < N_NODES) {
        float d = deg[i];
        deg[i] = (d > 0.0f) ? rsqrtf(d) : 0.0f;
    }
}

// ---------------- propagation ----------------
// self-loop term initializes h_next:  h_next[i] = dinv[i]^2 * h[i]
__global__ void k_self(const float* __restrict__ h, const float* __restrict__ dinv,
                       float* __restrict__ hn) {
    int t = blockIdx.x * blockDim.x + threadIdx.x;   // N * 16 float4 slots
    if (t < N_NODES * (FEAT / 4)) {
        int node = t >> 4;
        float s = dinv[node]; s *= s;
        float4 v = ((const float4*)h)[t];
        float4 o; o.x = v.x * s; o.y = v.y * s; o.z = v.z * s; o.w = v.w * s;
        ((float4*)hn)[t] = o;
    }
}

// one wave32 per edge, each lane handles 2 features
__global__ void k_scatter(const int* __restrict__ src, const int* __restrict__ dst,
                          const float* __restrict__ dinv,
                          const float* __restrict__ h, float* __restrict__ hn) {
    int t = blockIdx.x * blockDim.x + threadIdx.x;
    int e = t >> 5;
    int lane = t & 31;
    if (e < N_EDGES) {
        int s = src[e];
        int d = dst[e];
        float coef = dinv[s] * dinv[d];
        float2 v = ((const float2*)(h + (size_t)s * FEAT))[lane];
        float* o = hn + (size_t)d * FEAT + lane * 2;
        atomic_add_f32(o,     coef * v.x);
        atomic_add_f32(o + 1, coef * v.y);
    }
}

// ---------------- classifier: WMMA GEMM + log_softmax ----------------
// block = 128 threads = 4 waves; each wave handles 16 nodes.
// D = A(16x64 f16) x B(64x48 f16) accumulated f32, 3 N-tiles of 16, 2 K-chunks of 32.
__global__ void __launch_bounds__(128)
k_classifier(const float* __restrict__ h, const float* __restrict__ W,
             const float* __restrict__ b, float* __restrict__ out) {
    __shared__ float s_logits[4][16][49];   // padded: stride 49 coprime with 64 banks

    const int lane = threadIdx.x & 31;
    const int wave = threadIdx.x >> 5;
    const int g    = lane >> 4;     // half-wave group
    const int mn   = lane & 15;     // M (for A/D) or N (for B/D)
    const int nodeBase = blockIdx.x * 64 + wave * 16;

    // ---- A fragments (16-bit A 16x32 layout, ISA 7.12.2) ----
    int row = nodeBase + mn;
    if (row >= N_NODES) row = N_NODES - 1;          // clamp; output guarded below
    const float4* hr4 = (const float4*)(h + (size_t)row * FEAT);

    v16h a[2];
    #pragma unroll
    for (int kc = 0; kc < 2; ++kc) {
        const int q0 = (kc * 32 + 8 * g) >> 2;      // K = kc*32 + 8g + {0..7}
        float4 p0 = hr4[q0];
        float4 p1 = hr4[q0 + 1];
        float4 p2 = hr4[q0 + 4];                    // K = kc*32 + 16 + 8g + {0..7}
        float4 p3 = hr4[q0 + 5];
        a[kc][0]  = (_Float16)p0.x; a[kc][1]  = (_Float16)p0.y;
        a[kc][2]  = (_Float16)p0.z; a[kc][3]  = (_Float16)p0.w;
        a[kc][4]  = (_Float16)p1.x; a[kc][5]  = (_Float16)p1.y;
        a[kc][6]  = (_Float16)p1.z; a[kc][7]  = (_Float16)p1.w;
        a[kc][8]  = (_Float16)p2.x; a[kc][9]  = (_Float16)p2.y;
        a[kc][10] = (_Float16)p2.z; a[kc][11] = (_Float16)p2.w;
        a[kc][12] = (_Float16)p3.x; a[kc][13] = (_Float16)p3.y;
        a[kc][14] = (_Float16)p3.z; a[kc][15] = (_Float16)p3.w;
    }

    // ---- 3 N-tiles of 16 classes (40 padded to 48) ----
    #pragma unroll
    for (int t = 0; t < 3; ++t) {
        const int  cls   = t * 16 + mn;
        const bool valid = (cls < NCLS);
        const int  clsC  = valid ? cls : (NCLS - 1);           // always-valid address
        const float4* wr4 = (const float4*)(W + (size_t)clsC * FEAT);

        v16h bt[2];
        #pragma unroll
        for (int kc = 0; kc < 2; ++kc) {
            const int q0 = (kc * 32 + 8 * g) >> 2;             // B: elem j<8 <-> K = kc*32+8g+j, j>=8 <-> kc*32+16+8g+(j-8)
            #pragma unroll
            for (int q = 0; q < 4; ++q) {
                float4 w4 = wr4[q0 + q + (q >> 1) * 2];        // unconditional vector load
                bt[kc][q * 4 + 0] = (_Float16)(valid ? w4.x : 0.0f);  // cndmask, not branch
                bt[kc][q * 4 + 1] = (_Float16)(valid ? w4.y : 0.0f);
                bt[kc][q * 4 + 2] = (_Float16)(valid ? w4.z : 0.0f);
                bt[kc][q * 4 + 3] = (_Float16)(valid ? w4.w : 0.0f);
            }
        }

        float bv   = b[clsC];                                  // unconditional load
        float bias = valid ? bv : 0.0f;

        v8f acc = {};
        acc = __builtin_amdgcn_wmma_f32_16x16x32_f16(false, a[0], false, bt[0],
                                                     (short)0, acc, false, false);
        acc = __builtin_amdgcn_wmma_f32_16x16x32_f16(false, a[1], false, bt[1],
                                                     (short)0, acc, false, false);

        // D layout: lane -> N = mn, VGPR r -> M = r + 8g ; fold bias in here
        #pragma unroll
        for (int r = 0; r < 8; ++r)
            s_logits[wave][r + 8 * g][t * 16 + mn] = acc[r] + bias;
    }

    __syncthreads();

    // ---- per-node log_softmax over 40 classes (lanes 0..15 of each wave) ----
    if (g == 0) {
        int node = nodeBase + mn;
        if (node < N_NODES) {
            float mx = -3.402823e38f;
            #pragma unroll 8
            for (int c = 0; c < NCLS; ++c)
                mx = fmaxf(mx, s_logits[wave][mn][c]);
            float sum = 0.0f;
            #pragma unroll 8
            for (int c = 0; c < NCLS; ++c)
                sum += __expf(s_logits[wave][mn][c] - mx);
            float lse = mx + __logf(sum);
            float* op = out + (size_t)node * NCLS;
            #pragma unroll 8
            for (int c = 0; c < NCLS; ++c)
                op[c] = s_logits[wave][mn][c] - lse;
        }
    }
}

// ---------------- driver ----------------
extern "C" void kernel_launch(void* const* d_in, const int* in_sizes, int n_in,
                              void* d_out, int out_size, void* d_ws, size_t ws_size,
                              hipStream_t stream) {
    const float* x     = (const float*)d_in[0];
    const int*   edges = (const int*)d_in[1];
    const int*   src   = edges;               // edge_index[0]
    const int*   dst   = edges + N_EDGES;     // edge_index[1]
    const float* W     = (const float*)d_in[2];
    const float* b     = (const float*)d_in[3];
    float* out = (float*)d_out;

    float* h0   = (float*)d_ws;                       // N*64 f32
    float* h1   = h0 + (size_t)N_NODES * FEAT;        // N*64 f32
    float* dinv = h1 + (size_t)N_NODES * FEAT;        // N f32 (deg then dinv in place)

    const int TPB = 256;
    const int gN  = (N_NODES + TPB - 1) / TPB;
    const int gE  = (N_EDGES + TPB - 1) / TPB;
    const int gSelf = (N_NODES * (FEAT / 4) + TPB - 1) / TPB;
    const int gScat = (int)(((long long)N_EDGES * 32 + TPB - 1) / TPB);
    const int gCls  = (N_NODES + 63) / 64;

    // normalization
    k_init_deg<<<gN, TPB, 0, stream>>>(dinv);
    k_accum_deg<<<gE, TPB, 0, stream>>>(dst, dinv);
    k_dinv<<<gN, TPB, 0, stream>>>(dinv);

    // K = 3 propagation steps (ping-pong: x -> h0 -> h1 -> h0)
    k_self<<<gSelf, TPB, 0, stream>>>(x, dinv, h0);
    k_scatter<<<gScat, TPB, 0, stream>>>(src, dst, dinv, x, h0);

    k_self<<<gSelf, TPB, 0, stream>>>(h0, dinv, h1);
    k_scatter<<<gScat, TPB, 0, stream>>>(src, dst, dinv, h0, h1);

    k_self<<<gSelf, TPB, 0, stream>>>(h1, dinv, h0);
    k_scatter<<<gScat, TPB, 0, stream>>>(src, dst, dinv, h1, h0);

    // classifier + log_softmax
    k_classifier<<<gCls, 128, 0, stream>>>(h0, W, b, out);
}
